// DecoderRNN_9517647528119
// MI455X (gfx1250) — matrix operand order repaired and checked by reading, not compiled, not run on hardware
//
#include <hip/hip_runtime.h>

// Problem constants (match reference)
#define B_   1024
#define S_   256
#define V_   100
#define E_   128
#define H_   512
#define G3_  1536          // 3*H
#define VP_  112           // V padded to 7 WMMA N-tiles
#define ROWS 16            // batch rows per workgroup
#define NWG  (B_ / ROWS)   // 64 workgroups
#define HBS  520           // padded bf16 h-row stride (bank stagger)

typedef __attribute__((ext_vector_type(16))) __bf16 v16bf;
typedef __attribute__((ext_vector_type(8)))  __bf16 v8bf;
typedef __attribute__((ext_vector_type(8)))  float  v8f;

// Gate-triple column permutation: permuted column n' -> original W row.
//   jt = n'/48, g = (n'%48)/16, jj = n'%16  =>  orig row = g*H + jt*16 + jj
// So tiles 3*jt+0/1/2 hold gates r/z/n for hidden units j in [16*jt, 16*jt+16).

// ---------------- prep kernels (tiny, run once per launch) ----------------

// gi_perm[v][n'] = emb[v] . W_ih[orig(n')] + b_ih[orig] + (g<2 ? b_hh[orig] : 0)
__global__ void prep_gi_perm(const float* __restrict__ emb,
                             const float* __restrict__ W_ih,
                             const float* __restrict__ b_ih,
                             const float* __restrict__ b_hh,
                             float* __restrict__ gi_perm) {
  int t = blockIdx.x * blockDim.x + threadIdx.x;
  if (t >= V_ * G3_) return;
  int v = t / G3_, np = t - v * G3_;
  int jt = np / 48, rem = np - jt * 48, g = rem >> 4, jj = rem & 15;
  int jo = g * H_ + jt * 16 + jj;          // original row of W_ih / W_hh
  const float* er = emb + v * E_;
  const float* wr = W_ih + (size_t)jo * E_;
  float s = b_ih[jo] + (g < 2 ? b_hh[jo] : 0.0f);
#pragma unroll 8
  for (int e = 0; e < E_; ++e) s = fmaf(er[e], wr[e], s);
  gi_perm[t] = s;
}

// whh_perm[n'][k] = bf16( W_hh[orig(n')][k] )
__global__ void prep_whh_perm(const float* __restrict__ W_hh, __bf16* __restrict__ whh) {
  int t = blockIdx.x * blockDim.x + threadIdx.x;
  if (t >= G3_ * H_) return;
  int np = t / H_, k = t - np * H_;
  int jt = np / 48, rem = np - jt * 48, g = rem >> 4, jj = rem & 15;
  whh[t] = (__bf16)W_hh[(size_t)(g * H_ + jt * 16 + jj) * H_ + k];
}

__global__ void prep_wout_bf16(const float* __restrict__ W_out, __bf16* __restrict__ wout) {
  int t = blockIdx.x * blockDim.x + threadIdx.x;
  if (t >= VP_ * H_) return;
  int n = t / H_, k = t - n * H_;
  wout[t] = (__bf16)(n < V_ ? W_out[n * H_ + k] : 0.0f);
}

// ---------------- fragment helpers (CDNA5 wave32 WMMA layouts) ----------------

// A-fragment: 16x32 bf16 tile of h. lanes 0-15: M=lane, K k0..k0+15;
// lanes 16-31: M=lane-16, K k0+16..k0+31.
__device__ __forceinline__ v16bf load_a_frag(const __bf16* hrow_base, int lane, int k0) {
  const int half = lane >> 4;
  const int mr   = lane & 15;
  const __bf16* row = hrow_base + mr * HBS;
  v8bf lo = *(const v8bf*)(row + k0 + half * 16);
  v8bf hi = *(const v8bf*)(row + k0 + half * 16 + 8);
  return __builtin_shufflevector(lo, hi, 0, 1, 2, 3, 4, 5, 6, 7, 8, 9, 10, 11, 12, 13, 14, 15);
}

// B-fragment: 32x16 bf16 tile of W^T. Column n = n0+(lane&15); lanes 0-15 hold
// K k0..k0+15, lanes 16-31 hold k0+16..k0+31 -> 32 contiguous bytes of row n of W.
__device__ __forceinline__ v16bf load_b_frag(const __bf16* W, int n0, int lane, int k0) {
  const int col = n0 + (lane & 15);
  const int ko  = k0 + (lane >> 4) * 16;
  return *(const v16bf*)(W + (size_t)col * H_ + ko);
}

// C 16x16 f32: lanes 0-15 -> N=lane, M=i; lanes 16-31 -> N=lane-16, M=i+8.
__device__ __forceinline__ void store_c(float* dst, int stride, int n0, int lane, v8f c) {
  const int half = lane >> 4;
  const int col  = n0 + (lane & 15);
#pragma unroll
  for (int i = 0; i < 8; ++i) dst[(i + 8 * half) * stride + col] = c[i];
}

__device__ __forceinline__ float sigmoidf_(float x) { return 1.0f / (1.0f + __expf(-x)); }

// ---------------- main persistent GRU kernel ----------------

__global__ __launch_bounds__(256)
void decoder_rnn_kernel(const int*   __restrict__ inputs,   // (B,S)
                        const float* __restrict__ z,        // (B,H)
                        const float* __restrict__ b_hh,     // (3H) - only n-slice used here
                        const float* __restrict__ b_out,    // (V)
                        const float* __restrict__ gi_perm,  // (V,3H) permuted + biases folded
                        const __bf16* __restrict__ whh,     // (3H,H) bf16, row-permuted
                        const __bf16* __restrict__ wout,    // (VP,H) bf16
                        float* __restrict__ out,            // [0]=loss, [1..]=preds (B,S)
                        float* __restrict__ partials) {
  extern __shared__ unsigned char smem[];
  float*  hf     = (float*)smem;                   // ROWS*H_   fp32 hidden
  __bf16* hb     = (__bf16*)(hf + ROWS * H_);      // ROWS*HBS  bf16 hidden (padded)
  float*  lg     = (float*)(hb + ROWS * HBS);      // ROWS*VP_  logits
  float*  bhn    = lg + ROWS * VP_;                // H_  b_hh n-gate slice
  int*    tok    = (int*)(bhn + H_);               // ROWS
  int*    tgt    = tok + ROWS;                     // ROWS
  float*  nllrow = (float*)(tgt + ROWS);           // ROWS
  float*  nacc   = nllrow + ROWS;                  // 1

  const int tid  = threadIdx.x;
  const int lane = tid & 31;
  const int wave = tid >> 5;
  const int half = lane >> 4;
  const int jj   = lane & 15;
  const int b0   = blockIdx.x * ROWS;

  // h = z for this WG's 16 batch rows; stage b_hh n-slice
  for (int idx = tid; idx < ROWS * H_; idx += 256) {
    int m = idx >> 9, j = idx & (H_ - 1);
    float v = z[(size_t)(b0 + m) * H_ + j];
    hf[m * H_ + j]  = v;
    hb[m * HBS + j] = (__bf16)v;
  }
  for (int idx = tid; idx < H_; idx += 256) bhn[idx] = b_hh[2 * H_ + idx];
  if (tid == 0) *nacc = 0.0f;
  __syncthreads();

  for (int s = 0; s < S_; ++s) {
    // phase 0: decoder-input token (BOS=2 at s=0) and target token per row
    if (tid < ROWS) {
      int b = b0 + tid;
      tok[tid] = (s == 0) ? 2 : inputs[(size_t)b * S_ + s - 1];
      tgt[tid] = inputs[(size_t)b * S_ + s];
    }
    __syncthreads();

    // phase 1: gh = h @ W_hh^T with gate-triple tiling.
    // Wave owns j-triples jt0..jt0+3 (tiles 3*jt+{0,1,2} = r,z,n for same j's).
    {
      const int jt0 = wave * 4;
      v8f acc[12];
#pragma unroll
      for (int t = 0; t < 12; ++t) acc[t] = (v8f){0.f, 0.f, 0.f, 0.f, 0.f, 0.f, 0.f, 0.f};
#pragma unroll 1
      for (int kc = 0; kc < H_ / 32; ++kc) {
        const int k0 = kc * 32;
        v16bf a = load_a_frag(hb, lane, k0);
#pragma unroll
        for (int t = 0; t < 12; ++t) {
          const int n0 = jt0 * 48 + t * 16;   // 4 triples x 3 gate tiles, contiguous
          v16bf bf = load_b_frag(whh, n0, lane, k0);
          acc[t] = __builtin_amdgcn_wmma_f32_16x16x32_bf16(
              false, a, false, bf, (short)0, acc[t], false, false);
        }
      }
      __syncthreads();   // all reads of old hb complete before h is rewritten

      // GRU gates fully in registers: acc[3q+0]=r, acc[3q+1]=z, acc[3q+2]=n
      int tk[8];
#pragma unroll
      for (int i = 0; i < 8; ++i) tk[i] = tok[i + 8 * half];
#pragma unroll
      for (int q = 0; q < 4; ++q) {
        const int jt = jt0 + q;
        const int j  = jt * 16 + jj;
        const float bn = bhn[j];
#pragma unroll
        for (int i = 0; i < 8; ++i) {
          const int m = i + 8 * half;
          const float* gp = gi_perm + (size_t)tk[i] * G3_ + jt * 48;
          float gi_r = gp[jj];           // b_ih + b_hh folded in
          float gi_z = gp[16 + jj];      // b_ih + b_hh folded in
          float gi_n = gp[32 + jj];      // b_ih folded in
          float r  = sigmoidf_(acc[q * 3 + 0][i] + gi_r);
          float zg = sigmoidf_(acc[q * 3 + 1][i] + gi_z);
          float n  = tanhf(gi_n + r * (acc[q * 3 + 2][i] + bn));
          float hn = (1.0f - zg) * n + zg * hf[m * H_ + j];
          hf[m * H_ + j]  = hn;          // exclusive owner of (m,j)
          hb[m * HBS + j] = (__bf16)hn;
        }
      }
    }
    __syncthreads();

    // phase 2: logits = h_new @ W_out^T  (16 x 112), 7 N-tiles on waves 0..6
    if (wave < 7) {
      v8f acc = (v8f){0.f, 0.f, 0.f, 0.f, 0.f, 0.f, 0.f, 0.f};
      const int n0 = wave * 16;
#pragma unroll 1
      for (int kc = 0; kc < H_ / 32; ++kc) {
        const int k0 = kc * 32;
        v16bf a  = load_a_frag(hb, lane, k0);
        v16bf bf = load_b_frag(wout, n0, lane, k0);
        acc = __builtin_amdgcn_wmma_f32_16x16x32_bf16(
            false, a, false, bf, (short)0, acc, false, false);
      }
      store_c(lg, VP_, n0, lane, acc);
    } else {
      // idle wave: keep next-step W_hh panels warm (global_prefetch_b8)
      __builtin_prefetch((const void*)(whh + (size_t)lane * 24576), 0, 1);
    }
    __syncthreads();

    // phase 3: log-softmax / NLL / argmax — 16 lanes per batch row
    {
      const int m = tid >> 4, sub = tid & 15;
      float v[7];
      float mx = -1e30f;
      int   ai = V_;
#pragma unroll
      for (int i = 0; i < 7; ++i) {
        int j = sub + 16 * i;
        float val = (j < V_) ? (lg[m * VP_ + j] + b_out[j]) : -1e30f;
        v[i] = val;
        if (val > mx) { mx = val; ai = j; }   // ascending scan keeps first max
      }
#pragma unroll
      for (int off = 8; off >= 1; off >>= 1) {
        float om = __shfl_xor(mx, off, 16);
        int   oa = __shfl_xor(ai, off, 16);
        if (om > mx || (om == mx && oa < ai)) { mx = om; ai = oa; }
      }
      float se = 0.0f;
#pragma unroll
      for (int i = 0; i < 7; ++i) se += __expf(v[i] - mx);
#pragma unroll
      for (int off = 8; off >= 1; off >>= 1) se += __shfl_xor(se, off, 16);
      if (sub == 0) {
        int t = tgt[m];
        float logp_t = (lg[m * VP_ + t] + b_out[t]) - mx - __logf(se);
        nllrow[m] = -logp_t;
        out[1 + (size_t)(b0 + m) * S_ + s] = (float)ai;
      }
    }
    __syncthreads();

    // ordered (deterministic) per-step accumulation
    if (tid == 0) {
      float sacc = 0.0f;
#pragma unroll
      for (int m = 0; m < ROWS; ++m) sacc += nllrow[m];
      *nacc += sacc;
    }
  }

  __syncthreads();
  if (tid == 0) partials[blockIdx.x] = *nacc;
}

// ordered final reduction -> loss = sum_s mean_b nll = (1/B) * sum(all nll)
__global__ void finalize_loss(const float* __restrict__ partials, float* __restrict__ out) {
  if (blockIdx.x == 0 && threadIdx.x == 0) {
    float s = 0.0f;
    for (int i = 0; i < NWG; ++i) s += partials[i];
    out[0] = s * (1.0f / (float)B_);
  }
}

// ---------------- launch ----------------

extern "C" void kernel_launch(void* const* d_in, const int* in_sizes, int n_in,
                              void* d_out, int out_size, void* d_ws, size_t ws_size,
                              hipStream_t stream) {
  const int*   inputs = (const int*)d_in[0];
  const float* zin    = (const float*)d_in[1];
  const float* emb    = (const float*)d_in[2];
  const float* W_ih   = (const float*)d_in[3];
  const float* W_hh   = (const float*)d_in[4];
  const float* b_ih   = (const float*)d_in[5];
  const float* b_hh   = (const float*)d_in[6];
  const float* W_out  = (const float*)d_in[7];
  const float* b_out  = (const float*)d_in[8];
  float* out = (float*)d_out;

  char* ws = (char*)d_ws;
  float*  gi_perm  = (float*)ws;   ws += (size_t)V_ * G3_ * sizeof(float);
  __bf16* whh      = (__bf16*)ws;  ws += (size_t)G3_ * H_ * sizeof(__bf16);
  __bf16* wout     = (__bf16*)ws;  ws += (size_t)VP_ * H_ * sizeof(__bf16);
  float*  partials = (float*)ws;

  prep_gi_perm<<<(V_ * G3_ + 255) / 256, 256, 0, stream>>>(emb, W_ih, b_ih, b_hh, gi_perm);
  prep_whh_perm<<<(G3_ * H_ + 255) / 256, 256, 0, stream>>>(W_hh, whh);
  prep_wout_bf16<<<(VP_ * H_ + 255) / 256, 256, 0, stream>>>(W_out, wout);

  size_t smem = (size_t)ROWS * H_ * 4      // hf
              + (size_t)ROWS * HBS * 2     // hb
              + (size_t)ROWS * VP_ * 4     // lg
              + (size_t)H_ * 4             // bhn
              + (size_t)ROWS * 4 * 3       // tok, tgt, nllrow
              + 16;                        // nacc + slack
  decoder_rnn_kernel<<<NWG, 256, smem, stream>>>(inputs, zin, b_hh, b_out,
                                                 gi_perm, whh, wout, out, partials);
  finalize_loss<<<1, 1, 0, stream>>>(partials, out);
}
